// SelfAttentionTransformer_30786325578102
// MI455X (gfx1250) — compile-verified, matrix operand order checked by reading, not run on hardware
//
#include <hip/hip_runtime.h>
#include <math.h>

// ---------------------------------------------------------------------------
// Problem constants (from the reference): L=4, D=1024, H=8, T=1000, DFF=4096
// ---------------------------------------------------------------------------
#define kL   4
#define kD   1024
#define kH   8
#define kT   1000
#define kTP  1024      // T padded to a multiple of 32 (K-dim for WMMA)
#define kDFF 4096
#define kHS  128
#define kB   8
#define kM   (kB * kT) // 8000 rows of activations

typedef __attribute__((ext_vector_type(16))) __bf16 v16bf;
typedef __attribute__((ext_vector_type(8)))  float  v8f;

union Frag {
    uint4          q[2];   // two 16B chunks per lane
    v16bf          v;
    unsigned short s[16];
};

// round-to-nearest-even fp32 -> bf16 (bit pattern)
__device__ __forceinline__ unsigned short f2bf(float x) {
    union { float f; unsigned u; } c; c.f = x;
    unsigned r = c.u + 0x7FFFu + ((c.u >> 16) & 1u);
    return (unsigned short)(r >> 16);
}

__device__ __forceinline__ v8f wmma_bf16(v16bf a, v16bf b, v8f c) {
    // D = A(16x32 bf16) * B(32x16 bf16) + C(16x16 f32)
    return __builtin_amdgcn_wmma_f32_16x16x32_bf16(false, a, false, b,
                                                   (short)0, c, false, false);
}

// Load one 16x32-bf16 operand fragment: lane holds row (lane%16), two
// contiguous 8-element (16B) chunks at kbase+(lane/16)*8 and +16.
__device__ __forceinline__ void ld_frag(Frag& f, const unsigned short* p) {
    f.q[0] = *reinterpret_cast<const uint4*>(p);
    f.q[1] = *reinterpret_cast<const uint4*>(p + 16);
}

// CDNA5 async DMA: copy 16B from global to LDS, tracked by ASYNCcnt.
__device__ __forceinline__ void async_cp16(unsigned lds_off, const void* g) {
    asm volatile("global_load_async_to_lds_b128 %0, %1, off"
                 :: "v"(lds_off), "v"(g) : "memory");
}
__device__ __forceinline__ void wait_async0() {
    asm volatile("s_wait_asynccnt 0x0" ::: "memory");
}
__device__ __forceinline__ unsigned lds_addr(const void* p) {
    // generic pointer to LDS: low 32 bits are the LDS byte offset
    return (unsigned)(unsigned long long)p;
}

// ---------------------------------------------------------------------------
// Elementwise conversion / copy kernels
// ---------------------------------------------------------------------------
__global__ void copy_cvt_kernel(const float* __restrict__ in,
                                float* __restrict__ outf,
                                unsigned short* __restrict__ outb, int n) {
    int i = blockIdx.x * 256 + threadIdx.x;
    if (i >= n) return;
    float v = in[i];
    if (outf) outf[i] = v;
    if (outb) outb[i] = f2bf(v);
}

// [rows, kT] fp32 -> [rows, kTP] bf16, zero K-padding (for Wpk/Wpv)
__global__ void cvt_padk_kernel(const float* __restrict__ in,
                                unsigned short* __restrict__ out, int n) {
    int i = blockIdx.x * 256 + threadIdx.x;
    if (i >= n) return;
    int kp = i % kTP, r = i / kTP;
    out[i] = (kp < kT) ? f2bf(in[r * kT + kp]) : (unsigned short)0;
}

// [kB, kT, kD] fp32 -> [kB, kTP, kD] bf16, zero row padding
__global__ void cvt_rowpad_kernel(const float* __restrict__ in,
                                  unsigned short* __restrict__ out, int n) {
    int i = blockIdx.x * 256 + threadIdx.x;
    if (i >= n) return;
    int d = i % kD, t = (i / kD) % kTP, b = i / (kD * kTP);
    out[i] = (t < kT) ? f2bf(in[((size_t)b * kT + t) * kD + d])
                      : (unsigned short)0;
}

// ---------------------------------------------------------------------------
// RoPE (in place, fp32 [kB, kT, kD]); pairs (2i,2i+1) within each head
// ---------------------------------------------------------------------------
__global__ void rope_kernel(float* __restrict__ p, int n) {
    int idx = blockIdx.x * 256 + threadIdx.x;
    if (idx >= n) return;
    int pr = idx & 63;              // pair index 0..63
    int h  = (idx >> 6) & 7;        // head
    int r  = idx >> 9;
    int t  = r % kT;
    int b  = r / kT;
    float theta = __powf(10000.0f, -2.0f * (float)pr / (float)kHS);
    float ang = (float)t * theta;
    float c = cosf(ang), s = sinf(ang);
    size_t off = ((size_t)b * kT + t) * kD + h * kHS + pr * 2;
    float xr = p[off], xi = p[off + 1];
    p[off]     = xr * c - xi * s;
    p[off + 1] = xr * s + xi * c;
}

// ---------------------------------------------------------------------------
// LDS tile transpose: per batch b, fp32 [kT, kD] -> bf16 [kD, kTP] (zero pad)
// ---------------------------------------------------------------------------
__global__ void transpose_kernel(const float* __restrict__ in,
                                 unsigned short* __restrict__ out) {
    __shared__ float tile[32][33];
    int b = blockIdx.z;
    const float* ib = in + (size_t)b * kT * kD;
    unsigned short* ob = out + (size_t)b * kD * kTP;
    int t0 = blockIdx.x * 32, d0 = blockIdx.y * 32;
    int tx = threadIdx.x, ty = threadIdx.y;
#pragma unroll
    for (int i = 0; i < 4; ++i) {
        int t = t0 + ty + i * 8;
        tile[ty + i * 8][tx] = (t < kT) ? ib[(size_t)t * kD + d0 + tx] : 0.0f;
    }
    __syncthreads();
#pragma unroll
    for (int i = 0; i < 4; ++i) {
        int d = d0 + ty + i * 8;
        ob[(size_t)d * kTP + t0 + tx] = f2bf(tile[tx][ty + i * 8]);
    }
}

// ---------------------------------------------------------------------------
// Generic batched NT GEMM: C[m,n] = sum_k A[m,k] * B[n,k]  (bf16 in, f32 acc)
// Block = 256 threads = 8 waves; block tile 128x128; wave tile 64x32
// BIAS_MODE: 0 none, 1 per-N column, 2 per-M row
// ZPAD: store zeros for n in [Nvalid, Nstore)  (padded output columns)
// ---------------------------------------------------------------------------
template <int BIAS_MODE, bool GELU_, bool OUT_BF16, bool ZPAD>
__global__ __launch_bounds__(256) void gemm_nt_kernel(
    const unsigned short* __restrict__ A, const unsigned short* __restrict__ Bm,
    const float* __restrict__ bias, float* __restrict__ Cf,
    unsigned short* __restrict__ Cb, int M, int Nvalid, int Nstore, int K,
    int lda, int ldb, int ldc, long long aBS, long long bBS, long long cBS) {
    int z = blockIdx.z;
    A  += (size_t)z * aBS;
    Bm += (size_t)z * bBS;
    int lane = threadIdx.x & 31, wave = threadIdx.x >> 5;
    int lm = lane & 15, half = lane >> 4;
    int rowBase = blockIdx.x * 128 + (wave >> 2) * 64;
    int colBase = blockIdx.y * 128 + (wave & 3) * 32;
    int mrow[4], ncol[2];
#pragma unroll
    for (int i = 0; i < 4; ++i) {
        int r = rowBase + i * 16 + lm;
        mrow[i] = (r < M) ? r : (M - 1);           // clamp keeps EXEC uniform
    }
#pragma unroll
    for (int j = 0; j < 2; ++j) {
        int c = colBase + j * 16 + lm;
        ncol[j] = (c < Nvalid) ? c : (Nvalid - 1);
    }
    v8f acc[4][2] = {};
    for (int kb = 0; kb < K; kb += 32) {
        Frag a[4], bfr[2];
        int ko = kb + half * 8;
#pragma unroll
        for (int i = 0; i < 4; ++i) ld_frag(a[i], A + (size_t)mrow[i] * lda + ko);
#pragma unroll
        for (int j = 0; j < 2; ++j) ld_frag(bfr[j], Bm + (size_t)ncol[j] * ldb + ko);
        __builtin_prefetch(A + (size_t)mrow[0] * lda + ko + 32, 0, 0);
#pragma unroll
        for (int i = 0; i < 4; ++i)
#pragma unroll
            for (int j = 0; j < 2; ++j)
                acc[i][j] = wmma_bf16(a[i].v, bfr[j].v, acc[i][j]);
    }
    if (OUT_BF16) Cb += (size_t)z * cBS; else Cf += (size_t)z * cBS;
    // C layout: VGPR r holds M = tileM + r + half*8, N = tileN + lane%16
#pragma unroll
    for (int i = 0; i < 4; ++i) {
        int gmBase = rowBase + i * 16 + half * 8;
#pragma unroll
        for (int r = 0; r < 8; ++r) {
            int gm = gmBase + r;
            if (gm >= M) continue;
#pragma unroll
            for (int j = 0; j < 2; ++j) {
                int gn = colBase + j * 16 + lm;
                bool inN = gn < Nvalid;
                if (!ZPAD && !inN) continue;
                if (gn >= Nstore) continue;
                float v = acc[i][j][r];
                if (BIAS_MODE == 1 && inN) v += bias[gn];
                if (BIAS_MODE == 2) v += bias[gm];
                if (GELU_) v = 0.5f * v * (1.0f + erff(v * 0.70710678118654752f));
                if (ZPAD && !inN) v = 0.0f;
                if (OUT_BF16) Cb[(size_t)gm * ldc + gn] = f2bf(v);
                else          Cf[(size_t)gm * ldc + gn] = v;
            }
        }
    }
}

// ---------------------------------------------------------------------------
// Flash attention, S^T formulation, with CDNA5 async global->LDS staging.
//   S^T[j,i] = K2[j,:] . Q[i,:]     O^T[d,i] += V^T[d,:] . P^T[:,i]
// All 8 waves of a block share (b,h): K/V tiles (8KB each per 32 keys) are
// DMA'd into LDS once per block with GLOBAL_LOAD_ASYNC_TO_LDS_B128 and
// double-buffered so tile j+1 copies overlap tile j's 16 WMMAs.
// Q  : bf16 [kB, kTP, kD], K2 : bf16 [kB, kTP, kD], Vt : bf16 [kB, kD, kTP]
// Y  : f32  [kB, kT,  kD]
// ---------------------------------------------------------------------------
__global__ __launch_bounds__(256) void attn_kernel(
    const unsigned short* __restrict__ Q, const unsigned short* __restrict__ Kc,
    const unsigned short* __restrict__ Vt, float* __restrict__ Y) {
    __shared__ __attribute__((aligned(16))) unsigned short Ks[2][32 * 128];
    __shared__ __attribute__((aligned(16))) unsigned short Vs[2][128 * 32];
    const int b = blockIdx.z, h = blockIdx.y;
    const int tid = threadIdx.x;
    const int lane = tid & 31, wave = tid >> 5;
    const int lm = lane & 15, half = lane >> 4;
    const int ibase = (blockIdx.x * 8 + wave) * 16;

    // cooperative async stage of one 32-key tile: K 32x128, V^T 128x32 (bf16)
    auto stage = [&](int jb, int sel) {
        // K tile: 512 16B chunks; chunk c -> row j=c/16, 8-dim group c%16
#pragma unroll
        for (int c = tid; c < 512; c += 256) {
            int j = c >> 4, p = (c & 15) * 8;
            async_cp16(lds_addr(&Ks[sel][j * 128 + p]),
                       Kc + ((size_t)(b * kTP + jb + j)) * kD + h * kHS + p);
        }
        // V tile: 512 16B chunks; chunk c -> dim d=c/4, 8-key group c%4
#pragma unroll
        for (int c = tid; c < 512; c += 256) {
            int d = c >> 2, p = (c & 3) * 8;
            async_cp16(lds_addr(&Vs[sel][d * 32 + p]),
                       Vt + ((size_t)(b * kD + h * kHS + d)) * kTP + jb + p);
        }
    };

    // preload Q fragments (B-operand): lane's query column i = ibase + lm
    const unsigned short* qrow =
        Q + ((size_t)(b * kTP + ibase + lm)) * kD + h * kHS;
    Frag qf[4];
#pragma unroll
    for (int kc = 0; kc < 4; ++kc) ld_frag(qf[kc], qrow + kc * 32 + half * 8);

    v8f O[8] = {};
    float mst = -3.0e38f, lst = 0.0f;
    const float sc = 0.08838834764831845f;  // 1/sqrt(128)
    const int NT = kTP / 32;

    stage(0, 0);
    for (int it = 0; it < NT; ++it) {
        wait_async0();       // this wave's chunks of tile `it` have landed
        __syncthreads();     // everyone's chunks landed; prev buffer consumed
        if (it + 1 < NT) stage((it + 1) * 32, (it + 1) & 1);
        const unsigned short* ksb = &Ks[it & 1][0];
        const unsigned short* vsb = &Vs[it & 1][0];
        const int jb = it * 32;

        v8f S0 = {}, S1 = {};
#pragma unroll
        for (int kc = 0; kc < 4; ++kc) {
            Frag a0, a1;
            ld_frag(a0, ksb + (size_t)lm * 128 + kc * 32 + half * 8);
            ld_frag(a1, ksb + (size_t)(16 + lm) * 128 + kc * 32 + half * 8);
            S0 = wmma_bf16(a0.v, qf[kc].v, S0);
            S1 = wmma_bf16(a1.v, qf[kc].v, S1);
        }
        // scale + key-pad mask; key index: jb + r + half*8 (+16 for S1)
        float s0[8], s1[8], mx = -3.0e38f;
#pragma unroll
        for (int r = 0; r < 8; ++r) {
            int j0 = jb + r + half * 8;
            s0[r] = (j0 < kT)      ? S0[r] * sc : -3.0e38f;
            s1[r] = (j0 + 16 < kT) ? S1[r] * sc : -3.0e38f;
            mx = fmaxf(mx, fmaxf(s0[r], s1[r]));
        }
        mx = fmaxf(mx, __shfl_xor(mx, 16, 32));   // combine the two j-halves
        float mnew = fmaxf(mst, mx);
        float rescale = __expf(mst - mnew);
        // P^T tile -> B-operand fragment (pure in-lane repack)
        Frag pf;
        float ps = 0.0f;
#pragma unroll
        for (int r = 0; r < 8; ++r) {
            float e0 = __expf(s0[r] - mnew);
            float e1 = __expf(s1[r] - mnew);
            ps += e0 + e1;
            pf.s[r]     = f2bf(e0);
            pf.s[r + 8] = f2bf(e1);
        }
        ps += __shfl_xor(ps, 16, 32);
        lst = lst * rescale + ps;
        mst = mnew;
#pragma unroll
        for (int dt = 0; dt < 8; ++dt)
#pragma unroll
            for (int r = 0; r < 8; ++r) O[dt][r] *= rescale;
        // O^T[dtile] += V^T-frag x P^T-frag (V fragments from LDS)
#pragma unroll
        for (int dt = 0; dt < 8; ++dt) {
            Frag vf;
            ld_frag(vf, vsb + (size_t)(dt * 16 + lm) * 32 + half * 8);
            O[dt] = wmma_bf16(vf.v, pf.v, O[dt]);
        }
    }
    int i = ibase + lm;
    if (i < kT) {
        float inv = 1.0f / lst;
        float* yrow = Y + ((size_t)b * kT + i) * kD + h * kHS;
#pragma unroll
        for (int dt = 0; dt < 8; ++dt)
#pragma unroll
            for (int r = 0; r < 8; ++r)
                yrow[dt * 16 + r + half * 8] = O[dt][r] * inv;
    }
}

// ---------------------------------------------------------------------------
// Residual add + LayerNorm; writes fp32 and bf16 copies of the output row
// ---------------------------------------------------------------------------
__global__ __launch_bounds__(256) void add_ln_kernel(
    const float* __restrict__ resid, const float* __restrict__ delta,
    const float* __restrict__ g, const float* __restrict__ bt,
    float* __restrict__ outf, unsigned short* __restrict__ outb) {
    __shared__ float buf[kD];
    __shared__ float red[16];
    int row = blockIdx.x;
    const float* r0 = resid + (size_t)row * kD;
    const float* d0 = delta + (size_t)row * kD;
    float s = 0.0f, ss = 0.0f;
    for (int i = threadIdx.x; i < kD; i += 256) {
        float v = r0[i] + d0[i];
        buf[i] = v;
        s += v; ss += v * v;
    }
#pragma unroll
    for (int m = 16; m; m >>= 1) {
        s  += __shfl_xor(s, m, 32);
        ss += __shfl_xor(ss, m, 32);
    }
    int wv = threadIdx.x >> 5;
    if ((threadIdx.x & 31) == 0) { red[wv] = s; red[wv + 8] = ss; }
    __syncthreads();
    if (threadIdx.x == 0) {
        float a = 0, c = 0;
        for (int i = 0; i < 8; ++i) { a += red[i]; c += red[i + 8]; }
        red[0] = a; red[8] = c;
    }
    __syncthreads();
    float mean = red[0] / (float)kD;
    float var  = red[8] / (float)kD - mean * mean;
    float rstd = rsqrtf(var + 1e-5f);
    for (int i = threadIdx.x; i < kD; i += 256) {
        float v = (buf[i] - mean) * rstd * g[i] + bt[i];
        outf[(size_t)row * kD + i] = v;
        outb[(size_t)row * kD + i] = f2bf(v);
    }
}

// ---------------------------------------------------------------------------
// Host orchestration
// ---------------------------------------------------------------------------
static inline int cdiv(int a, int b) { return (a + b - 1) / b; }

extern "C" void kernel_launch(void* const* d_in, const int* in_sizes, int n_in,
                              void* d_out, int out_size, void* d_ws,
                              size_t ws_size, hipStream_t stream) {
    (void)in_sizes; (void)n_in; (void)out_size;
    const float* x    = (const float*)d_in[0];
    const float* Wq   = (const float*)d_in[1];
    const float* Wk   = (const float*)d_in[2];
    const float* Wv   = (const float*)d_in[3];
    const float* Wpk  = (const float*)d_in[4];
    const float* bpk  = (const float*)d_in[5];
    const float* Wpv  = (const float*)d_in[6];
    const float* bpv  = (const float*)d_in[7];
    const float* ln1g = (const float*)d_in[8];
    const float* ln1b = (const float*)d_in[9];
    const float* W1   = (const float*)d_in[10];
    const float* b1   = (const float*)d_in[11];
    const float* W2   = (const float*)d_in[12];
    const float* b2   = (const float*)d_in[13];
    const float* ln2g = (const float*)d_in[14];
    const float* ln2b = (const float*)d_in[15];

    size_t off = 0;
    auto alloc = [&](size_t bytes) -> void* {
        void* p = (char*)d_ws + off;
        off += (bytes + 255) & ~(size_t)255;
        return p;
    };
    unsigned short* WqB  = (unsigned short*)alloc((size_t)kL * kD * kD * 2);
    unsigned short* WkB  = (unsigned short*)alloc((size_t)kL * kD * kD * 2);
    unsigned short* WvB  = (unsigned short*)alloc((size_t)kL * kD * kD * 2);
    unsigned short* WpkB = (unsigned short*)alloc((size_t)kL * kT * kTP * 2);
    unsigned short* WpvB = (unsigned short*)alloc((size_t)kL * kT * kTP * 2);
    unsigned short* W1B  = (unsigned short*)alloc((size_t)kL * kDFF * kD * 2);
    unsigned short* W2B  = (unsigned short*)alloc((size_t)kL * kD * kDFF * 2);
    float* cur = (float*)alloc((size_t)kM * kD * 4);
    float* qf  = (float*)alloc((size_t)kM * kD * 4);   // q, later v, scratch
    float* kf  = (float*)alloc((size_t)kM * kD * 4);   // k, later k2
    float* yb  = (float*)alloc((size_t)kM * kD * 4);   // attn out / ffn out
    unsigned short* curB = (unsigned short*)alloc((size_t)kM * kD * 2);
    unsigned short* qB   = (unsigned short*)alloc((size_t)kB * kTP * kD * 2);
    unsigned short* k2B  = (unsigned short*)alloc((size_t)kB * kTP * kD * 2);
    unsigned short* v2tB = (unsigned short*)alloc((size_t)kB * kD * kTP * 2);
    unsigned short* ttB  = (unsigned short*)alloc((size_t)kB * kD * kTP * 2);
    unsigned short* hB   = (unsigned short*)alloc((size_t)kM * kDFF * 2);
    if (off > ws_size) return;  // workspace too small: bail deterministically

    // --- one-time per call: weight conversion to bf16 (fits in 192MB L2) ---
    { int n = kL * kD * kD;
      copy_cvt_kernel<<<cdiv(n, 256), 256, 0, stream>>>(Wq, nullptr, WqB, n);
      copy_cvt_kernel<<<cdiv(n, 256), 256, 0, stream>>>(Wk, nullptr, WkB, n);
      copy_cvt_kernel<<<cdiv(n, 256), 256, 0, stream>>>(Wv, nullptr, WvB, n); }
    { int n = kL * kDFF * kD;
      copy_cvt_kernel<<<cdiv(n, 256), 256, 0, stream>>>(W1, nullptr, W1B, n);
      copy_cvt_kernel<<<cdiv(n, 256), 256, 0, stream>>>(W2, nullptr, W2B, n); }
    { int n = kL * kT * kTP;
      cvt_padk_kernel<<<cdiv(n, 256), 256, 0, stream>>>(Wpk, WpkB, n);
      cvt_padk_kernel<<<cdiv(n, 256), 256, 0, stream>>>(Wpv, WpvB, n); }
    { int n = kM * kD;
      copy_cvt_kernel<<<cdiv(n, 256), 256, 0, stream>>>(x, cur, curB, n); }

    const int nR  = kB * kT * kH * (kHS / 2);  // rope pairs
    const int nQP = kB * kTP * kD;
    dim3 blk(256);
    dim3 gProj(cdiv(kM, 128), kD / 128, 1);
    dim3 gT(kTP / 32, kD / 32, kB), bT(32, 8);
    dim3 gM1(cdiv(kT, 128), kD / 128, kB);
    dim3 gM2(kD / 128, kTP / 128, kB);
    dim3 gA(8, kH, kB);
    dim3 gF1(cdiv(kM, 128), kDFF / 128, 1);
    dim3 gF2(cdiv(kM, 128), kD / 128, 1);

    for (int l = 0; l < kL; ++l) {
        // Q = cur @ Wq^T ; K = cur @ Wk^T
        gemm_nt_kernel<0, false, false, false><<<gProj, blk, 0, stream>>>(
            curB, WqB + (size_t)l * kD * kD, nullptr, qf, nullptr,
            kM, kD, kD, kD, kD, kD, kD, 0, 0, 0);
        gemm_nt_kernel<0, false, false, false><<<gProj, blk, 0, stream>>>(
            curB, WkB + (size_t)l * kD * kD, nullptr, kf, nullptr,
            kM, kD, kD, kD, kD, kD, kD, 0, 0, 0);
        // RoPE(q), then bf16 row-padded copy for attention
        rope_kernel<<<cdiv(nR, 256), 256, 0, stream>>>(qf, nR);
        cvt_rowpad_kernel<<<cdiv(nQP, 256), 256, 0, stream>>>(qf, qB, nQP);
        // V = cur @ Wv^T (reuse qf)
        gemm_nt_kernel<0, false, false, false><<<gProj, blk, 0, stream>>>(
            curB, WvB + (size_t)l * kD * kD, nullptr, qf, nullptr,
            kM, kD, kD, kD, kD, kD, kD, 0, 0, 0);
        // k^T (bf16, padded) then k2[u,d] = Wpk[u,:] . kT[d,:] + bpk[u]
        transpose_kernel<<<gT, bT, 0, stream>>>(kf, ttB);
        gemm_nt_kernel<2, false, false, false><<<gM1, blk, 0, stream>>>(
            WpkB + (size_t)l * kT * kTP, ttB, bpk + (size_t)l * kT, kf, nullptr,
            kT, kD, kD, kTP, kTP, kTP, kD,
            0, (long long)kD * kTP, (long long)kT * kD);
        rope_kernel<<<cdiv(nR, 256), 256, 0, stream>>>(kf, nR);
        cvt_rowpad_kernel<<<cdiv(nQP, 256), 256, 0, stream>>>(kf, k2B, nQP);
        // v^T then v2^T[d,u] = vT[d,:] . Wpv[u,:] + bpv[u]  (bf16, zero-pad u)
        transpose_kernel<<<gT, bT, 0, stream>>>(qf, ttB);
        gemm_nt_kernel<1, false, true, true><<<gM2, blk, 0, stream>>>(
            ttB, WpvB + (size_t)l * kT * kTP, bpv + (size_t)l * kT,
            nullptr, v2tB, kD, kT, kTP, kTP, kTP, kTP, kTP,
            (long long)kD * kTP, 0, (long long)kD * kTP);
        // fused flash attention (async double-buffered K/V staging in LDS)
        attn_kernel<<<gA, blk, 0, stream>>>(qB, k2B, v2tB, yb);
        // residual + LN1
        add_ln_kernel<<<kM, blk, 0, stream>>>(cur, yb, ln1g + (size_t)l * kD,
                                              ln1b + (size_t)l * kD, cur, curB);
        // FFN
        gemm_nt_kernel<1, true, true, false><<<gF1, blk, 0, stream>>>(
            curB, W1B + (size_t)l * kDFF * kD, b1 + (size_t)l * kDFF,
            nullptr, hB, kM, kDFF, kDFF, kD, kD, kD, kDFF, 0, 0, 0);
        gemm_nt_kernel<1, false, false, false><<<gF2, blk, 0, stream>>>(
            hB, W2B + (size_t)l * kD * kDFF, b2 + (size_t)l * kD,
            yb, nullptr, kM, kD, kD, kDFF, kDFF, kDFF, kD, 0, 0, 0);
        // residual + LN2
        add_ln_kernel<<<kM, blk, 0, stream>>>(cur, yb, ln2g + (size_t)l * kD,
                                              ln2b + (size_t)l * kD, cur, curB);
    }
    // final output
    { int n = kM * kD;
      copy_cvt_kernel<<<cdiv(n, 256), 256, 0, stream>>>(cur, (float*)d_out,
                                                        nullptr, n); }
}
